// Mutual_Cross_Agent_Attention_75419625718308
// MI455X (gfx1250) — matrix operand order repaired and checked by reading, not compiled, not run on hardware
//
#include <hip/hip_runtime.h>

typedef __bf16 bf16_t;
typedef __attribute__((ext_vector_type(16))) __bf16 v16bf;
typedef __attribute__((ext_vector_type(8)))  __bf16 v8bf;
typedef __attribute__((ext_vector_type(8)))  float   v8f;

namespace {

constexpr int  kB   = 32;
constexpr int  kN   = 784;   // 28*28 tokens
constexpr int  kC   = 768;
constexpr int  kH   = 12;    // heads
constexpr int  kD   = 64;    // head dim
constexpr int  kA   = 49;    // agents
constexpr int  kAP  = 64;    // agents padded to WMMA tile
constexpr int  kNP  = 800;   // tokens padded to K%32==0
constexpr int  kHW  = 28;
constexpr long kBN  = (long)kB * kN;   // 25088
constexpr float kScale = 0.125f;       // 64^-0.5

__device__ __forceinline__ float waveMax(float v){
  for (int o = 16; o; o >>= 1) v = fmaxf(v, __shfl_xor(v, o, 32));
  return v;
}
__device__ __forceinline__ float waveSum(float v){
  for (int o = 16; o; o >>= 1) v += __shfl_xor(v, o, 32);
  return v;
}

// ---------------------------------------------------------------------------
// Compile-time-NSUB tile body: one 16x(NSUB*16) output tile per wave.
// All accumulator indexing is static -> no movrel; all control flow around the
// WMMA chain is scalar -> EXEC stays all-ones as the ISA requires.
// ---------------------------------------------------------------------------
template<int NSUB, bool OUT_BF16, bool TRANS>
__device__ __forceinline__
void gemm_tile_body(const bf16_t* __restrict__ A, long ldA,
                    const bf16_t* __restrict__ BT, long ldB,
                    void* __restrict__ Cv, long ldC,
                    const float* __restrict__ bias, long cOff,
                    int K, int tm, int col0, float alpha, int l16, int hp)
{
  // A fragment: lanes 0-15 hold row tm*16+l16, K = {0..7, 16..23}+k0
  //             lanes 16-31 hold same rows,   K = {8..15,24..31}+k0
  const bf16_t* arow = A + (long)(tm * 16 + l16) * ldA;
  // B fragment: lane holds output col (col0 + s*16 + l16), K = {0..7, 16..23}+hp*8+k0
  const bf16_t* brow = BT + (long)(col0 + l16) * ldB + hp * 8;

  v8f acc[NSUB];
#pragma unroll
  for (int s = 0; s < NSUB; ++s)
#pragma unroll
    for (int r = 0; r < 8; ++r) acc[s][r] = 0.f;

  for (int k0 = 0; k0 < K; k0 += 32) {
    v16bf afrag;
    const bf16_t* ap = arow + k0 + hp * 8;
    ((v8bf*)&afrag)[0] = *(const v8bf*)(ap);
    ((v8bf*)&afrag)[1] = *(const v8bf*)(ap + 16);
#pragma unroll
    for (int s = 0; s < NSUB; ++s) {
      const bf16_t* bp = brow + (long)s * 16 * ldB + k0;
      v16bf bfrag;
      ((v8bf*)&bfrag)[0] = *(const v8bf*)(bp);
      ((v8bf*)&bfrag)[1] = *(const v8bf*)(bp + 16);
      acc[s] = __builtin_amdgcn_wmma_f32_16x16x32_bf16(
          false, afrag, false, bfrag, (short)0, acc[s], false, false);
    }
  }

  // C layout: VGPR r -> (M = tm*16 + hp*8 + r, N = col0 + s*16 + l16)
  const int rbase = tm * 16 + hp * 8;
  if constexpr (OUT_BF16) {
    bf16_t* C = (bf16_t*)Cv + cOff;
#pragma unroll
    for (int s = 0; s < NSUB; ++s) {
      const int col = col0 + s * 16 + l16;
#pragma unroll
      for (int r = 0; r < 8; ++r) {
        float v = acc[s][r] * alpha;
        if (bias) v += bias[col];
        const long idx = TRANS ? ((long)col * ldC + (rbase + r))
                               : ((long)(rbase + r) * ldC + col);
        C[idx] = (bf16_t)v;
      }
    }
  } else {
    float* C = (float*)Cv + cOff;
#pragma unroll
    for (int s = 0; s < NSUB; ++s) {
      const int col = col0 + s * 16 + l16;
#pragma unroll
      for (int r = 0; r < 8; ++r) {
        float v = acc[s][r] * alpha;
        if (bias) v += bias[col];
        const long idx = TRANS ? ((long)col * ldC + (rbase + r))
                               : ((long)(rbase + r) * ldC + col);
        C[idx] = v;
      }
    }
  }
}

// ---------------------------------------------------------------------------
// Generic batched GEMM: C[M,Nc] = alpha * A[M,K] * B[K,Nc] (+ bias[col])
// A row-major bf16, B given as BT[Nc,K] row-major bf16 (so lane loads of both
// fragments are 16B-contiguous). Wave32: each wave owns a 16x64 tile (4 WMMA
// accumulators sharing one A fragment). K % 32 == 0, M % 16 == 0, Nc % 16 == 0.
// Batched over blockIdx.z = b*nH + h with independent strides per matrix.
// ---------------------------------------------------------------------------
template<bool OUT_BF16, bool TRANS>
__global__ __launch_bounds__(256)
void gemm_bf16_wmma(const bf16_t* __restrict__ A, long ldA,
                    const bf16_t* __restrict__ BT, long ldB,
                    void* __restrict__ Cv, long ldC,
                    const float* __restrict__ bias,
                    int M, int Nc, int K, int nH,
                    long aSB, long aSH, long bSB, long bSH, long cSB, long cSH,
                    float alpha)
{
  const int z  = blockIdx.z;
  const int bb = z / nH;
  const int hh = z - bb * nH;
  A  += (long)bb * aSB + (long)hh * aSH;
  BT += (long)bb * bSB + (long)hh * bSH;
  const long cOff = (long)bb * cSB + (long)hh * cSH;

  const int lane = threadIdx.x & 31;
  const int l16  = lane & 15;
  const int hp   = lane >> 4;
  // force wave id into an SGPR so tile selection is scalar (no exec masking)
  const int wave = __builtin_amdgcn_readfirstlane((int)(threadIdx.x >> 5));

  const int tilesN4 = (Nc + 63) >> 6;
  const long wid = (long)blockIdx.x * 8 + wave;
  if (wid >= (long)(M >> 4) * tilesN4) return;   // scalar branch, EXEC intact
  const int tm   = (int)(wid / tilesN4);
  const int col0 = (int)(wid % tilesN4) * 64;
  const int rem  = (Nc - col0) >> 4;
  const int nSub = rem > 4 ? 4 : rem;

  switch (nSub) {
    case 4: gemm_tile_body<4, OUT_BF16, TRANS>(A, ldA, BT, ldB, Cv, ldC, bias, cOff,
                                               K, tm, col0, alpha, l16, hp); break;
    case 3: gemm_tile_body<3, OUT_BF16, TRANS>(A, ldA, BT, ldB, Cv, ldC, bias, cOff,
                                               K, tm, col0, alpha, l16, hp); break;
    case 2: gemm_tile_body<2, OUT_BF16, TRANS>(A, ldA, BT, ldB, Cv, ldC, bias, cOff,
                                               K, tm, col0, alpha, l16, hp); break;
    default: gemm_tile_body<1, OUT_BF16, TRANS>(A, ldA, BT, ldB, Cv, ldC, bias, cOff,
                                                K, tm, col0, alpha, l16, hp); break;
  }
}

// ---------------------------------------------------------------------------
// Elementwise / staging kernels
// ---------------------------------------------------------------------------
__global__ void k_cvt_bf16(const float* __restrict__ in, bf16_t* __restrict__ out, long n) {
  long i = (long)blockIdx.x * blockDim.x + threadIdx.x;
  if (i < n) out[i] = (bf16_t)in[i];
}

// W[Kd,Nd] f32 -> WT[Nd,Kd] bf16
__global__ void k_transpose_w(const float* __restrict__ W, bf16_t* __restrict__ WT,
                              int Kd, int Nd) {
  long i = (long)blockIdx.x * blockDim.x + threadIdx.x;
  if (i >= (long)Kd * Nd) return;
  int n = (int)(i % Nd);
  int k = (int)(i / Nd);
  WT[(long)n * Kd + k] = (bf16_t)W[i];
}

// 4x4 avg-pool of q (bf16 [B,784,768]) -> padded agents ah (bf16 [B,12,64,64])
__global__ void k_pool_ah(const bf16_t* __restrict__ q, bf16_t* __restrict__ ah) {
  long i = (long)blockIdx.x * blockDim.x + threadIdx.x;
  if (i >= (long)kB * kAP * kC) return;
  int c = (int)(i % kC);
  int a = (int)((i / kC) % kAP);
  int b = (int)(i / ((long)kAP * kC));
  float v = 0.f;
  if (a < kA) {
    int ay = a / 7, ax = a % 7;
    for (int dy = 0; dy < 4; ++dy)
      for (int dx = 0; dx < 4; ++dx) {
        int n = (ay * 4 + dy) * kHW + (ax * 4 + dx);
        v += (float)q[((long)b * kN + n) * kC + c];
      }
    v *= (1.f / 16.f);
  }
  int h = c >> 6, d = c & 63;
  ah[(((long)b * kH + h) * kAP + a) * kD + d] = (bf16_t)v;
}

// V part of kv (bf16 [B,784,1536]) -> vT (bf16 [B,12,64,800]) zero-padded in n
__global__ void k_build_vT(const bf16_t* __restrict__ kv, bf16_t* __restrict__ vT) {
  long i = (long)blockIdx.x * blockDim.x + threadIdx.x;
  if (i >= (long)kB * kH * kD * kNP) return;
  int n = (int)(i % kNP);
  long r = i / kNP;
  int d = (int)(r % kD); r /= kD;
  int h = (int)(r % kH);
  int b = (int)(r / kH);
  float v = 0.f;
  if (n < kN) v = (float)kv[((long)b * kN + n) * (2 * kC) + kC + h * kD + d];
  vT[i] = (bf16_t)v;
}

// half-pixel bilinear sample of a 7x7 image at output pixel (y,x) of 28x28
__device__ __forceinline__ float bilerp7(const float* __restrict__ img, int y, int x) {
  float sy = (y + 0.5f) * 0.25f - 0.5f;
  float sx = (x + 0.5f) * 0.25f - 0.5f;
  int y0 = (int)floorf(sy); float wy = sy - (float)y0;
  int x0 = (int)floorf(sx); float wx = sx - (float)x0;
  int y0c = min(max(y0, 0), 6), y1c = min(max(y0 + 1, 0), 6);
  int x0c = min(max(x0, 0), 6), x1c = min(max(x0 + 1, 0), 6);
  float a = img[y0c * 7 + x0c], b = img[y0c * 7 + x1c];
  float c = img[y1c * 7 + x0c], d = img[y1c * 7 + x1c];
  return a * (1.f - wy) * (1.f - wx) + b * (1.f - wy) * wx
       + c * wy * (1.f - wx) + d * wy * wx;
}

// pos_bias[h,a,n] = resize(an)[h,a,y,x] + ah_bias[h,a,y] + aw_bias[h,a,x]
__global__ void k_pos_bias(const float* __restrict__ an, const float* __restrict__ ahb,
                           const float* __restrict__ awb, float* __restrict__ pb) {
  long i = (long)blockIdx.x * blockDim.x + threadIdx.x;
  if (i >= (long)kH * kA * kN) return;
  int n = (int)(i % kN);
  int a = (int)((i / kN) % kA);
  int h = (int)(i / ((long)kA * kN));
  int y = n / kHW, x = n % kHW;
  pb[i] = bilerp7(an + ((long)h * kA + a) * 49, y, x)
        + ahb[((long)h * kA + a) * kHW + y]
        + awb[((long)h * kA + a) * kHW + x];
}

// ag_bias[h,n,a] = resize(na)[h,a,y,x] + ha_bias[h,y,a] + wa_bias[h,x,a]
__global__ void k_ag_bias(const float* __restrict__ na, const float* __restrict__ hab,
                          const float* __restrict__ wab, float* __restrict__ ab) {
  long i = (long)blockIdx.x * blockDim.x + threadIdx.x;
  if (i >= (long)kH * kN * kA) return;
  int a = (int)(i % kA);
  int n = (int)((i / kA) % kN);
  int h = (int)(i / ((long)kN * kA));
  int y = n / kHW, x = n % kHW;
  ab[i] = bilerp7(na + ((long)h * kA + a) * 49, y, x)
        + hab[((long)h * kHW + y) * kA + a]
        + wab[((long)h * kHW + x) * kA + a];
}

// softmax over 784 tokens: probs[b,h,a,0:800] (bf16, tail & padded agents = 0)
__global__ __launch_bounds__(128)
void k_softmax_agent(const float* __restrict__ scores,  // [B,12,64,784] pre-scaled
                     const float* __restrict__ pb,      // [12,49,784]
                     bf16_t* __restrict__ probs) {      // [B,12,64,800]
  const int lane = threadIdx.x & 31;
  const long row = (long)blockIdx.x * 4 + (threadIdx.x >> 5);
  if (row >= (long)kB * kH * kAP) return;
  const int a = (int)(row % kAP);
  const int h = (int)((row / kAP) % kH);
  bf16_t* prow = probs + row * kNP;
  if (a >= kA) {
    for (int i = lane; i < kNP; i += 32) prow[i] = (bf16_t)0.f;
    return;
  }
  const float* srow = scores + row * kN;
  const float* brow = pb + ((long)h * kA + a) * kN;
  float buf[25];
  int cnt = 0;
  float mx = -3e38f;
  for (int i = lane; i < kN; i += 32) {
    float v = srow[i] + brow[i];
    buf[cnt++] = v;
    mx = fmaxf(mx, v);
  }
  mx = waveMax(mx);
  float sum = 0.f;
  for (int j = 0; j < cnt; ++j) { buf[j] = expf(buf[j] - mx); sum += buf[j]; }
  sum = waveSum(sum);
  const float inv = 1.f / sum;
  cnt = 0;
  for (int i = lane; i < kN; i += 32) prow[i] = (bf16_t)(buf[cnt++] * inv);
  for (int i = kN + lane; i < kNP; i += 32) prow[i] = (bf16_t)0.f;
}

// softmax over 49 agents: p2[b,h,n,0:64] (bf16, padded agents = 0)
__global__ __launch_bounds__(128)
void k_softmax_q(const float* __restrict__ s2,   // [B,12,784,64] pre-scaled
                 const float* __restrict__ ab,   // [12,784,49]
                 bf16_t* __restrict__ p2) {      // [B,12,784,64]
  const int lane = threadIdx.x & 31;
  const long row = (long)blockIdx.x * 4 + (threadIdx.x >> 5);
  if (row >= (long)kB * kH * kN) return;
  const int n = (int)(row % kN);
  const int h = (int)((row / kN) % kH);
  const float* srow = s2 + row * kAP;
  const float* brow = ab + ((long)h * kN + n) * kA;
  const int a0 = lane, a1 = lane + 32;
  float v0 = -3e38f, v1 = -3e38f;
  if (a0 < kA) v0 = srow[a0] + brow[a0];
  if (a1 < kA) v1 = srow[a1] + brow[a1];
  float mx = waveMax(fmaxf(v0, v1));
  float e0 = (a0 < kA) ? expf(v0 - mx) : 0.f;
  float e1 = (a1 < kA) ? expf(v1 - mx) : 0.f;
  float inv = 1.f / waveSum(e0 + e1);
  bf16_t* prow = p2 + row * kAP;
  prow[a0] = (bf16_t)(e0 * inv);
  prow[a1] = (bf16_t)(e1 * inv);
}

// out = bf16( outPre + depthwise3x3(v) + dwc_bias )
__global__ void k_dwc_add(const bf16_t* __restrict__ kv,    // [B,784,1536], V at +768
                          const float* __restrict__ w,      // [3,3,1,768]
                          const float* __restrict__ dwb,    // [768]
                          const float* __restrict__ outPre, // [B,784,768]
                          bf16_t* __restrict__ outBF) {     // [B,784,768]
  long i = (long)blockIdx.x * blockDim.x + threadIdx.x;
  if (i >= (long)kB * kN * kC) return;
  int c = (int)(i % kC);
  long bn = i / kC;
  int n = (int)(bn % kN);
  int b = (int)(bn / kN);
  int y = n / kHW, x = n % kHW;
  float s = dwb[c];
  for (int dy = -1; dy <= 1; ++dy)
    for (int dx = -1; dx <= 1; ++dx) {
      int yy = y + dy, xx = x + dx;
      if (yy >= 0 && yy < kHW && xx >= 0 && xx < kHW) {
        s += (float)kv[((long)b * kN + yy * kHW + xx) * (2 * kC) + kC + c]
           * w[((dy + 1) * 3 + (dx + 1)) * kC + c];
      }
    }
  outBF[i] = (bf16_t)(outPre[i] + s);
}

// host-side dispatch helper
static void run_gemm(hipStream_t stream,
                     const bf16_t* A, long ldA, const bf16_t* BT, long ldB,
                     void* C, long ldC, const float* bias,
                     int M, int Nc, int K, int nH,
                     long aSB, long aSH, long bSB, long bSH, long cSB, long cSH,
                     float alpha, int batches, bool outBF, bool trans) {
  long waves = (long)(M / 16) * ((Nc + 63) / 64);
  dim3 grid((unsigned)((waves + 7) / 8), 1, (unsigned)batches);
  dim3 blk(256, 1, 1);
  if (outBF && trans)
    gemm_bf16_wmma<true, true><<<grid, blk, 0, stream>>>(A, ldA, BT, ldB, C, ldC, bias,
        M, Nc, K, nH, aSB, aSH, bSB, bSH, cSB, cSH, alpha);
  else if (outBF)
    gemm_bf16_wmma<true, false><<<grid, blk, 0, stream>>>(A, ldA, BT, ldB, C, ldC, bias,
        M, Nc, K, nH, aSB, aSH, bSB, bSH, cSB, cSH, alpha);
  else
    gemm_bf16_wmma<false, false><<<grid, blk, 0, stream>>>(A, ldA, BT, ldB, C, ldC, bias,
        M, Nc, K, nH, aSB, aSH, bSB, bSH, cSB, cSH, alpha);
}

static inline dim3 ew(long n) { return dim3((unsigned)((n + 255) / 256)); }

} // namespace

extern "C" void kernel_launch(void* const* d_in, const int* in_sizes, int n_in,
                              void* d_out, int out_size, void* d_ws, size_t ws_size,
                              hipStream_t stream) {
  (void)in_sizes; (void)n_in; (void)out_size; (void)ws_size;
  // inputs (setup_inputs order); index 0 = rgb, 1 = depth
  const float* fea[2]    = {(const float*)d_in[0],  (const float*)d_in[1]};
  const float* q_w[2]    = {(const float*)d_in[2],  (const float*)d_in[6]};
  const float* kv_w[2]   = {(const float*)d_in[3],  (const float*)d_in[7]};
  const float* proj_w[2] = {(const float*)d_in[4],  (const float*)d_in[8]};
  const float* proj_b[2] = {(const float*)d_in[5],  (const float*)d_in[9]};
  const float* dwc_w[2]  = {(const float*)d_in[10], (const float*)d_in[12]};
  const float* dwc_b[2]  = {(const float*)d_in[11], (const float*)d_in[13]};
  const float* an_b[2]   = {(const float*)d_in[14], (const float*)d_in[20]};
  const float* na_b[2]   = {(const float*)d_in[15], (const float*)d_in[21]};
  const float* ah_b[2]   = {(const float*)d_in[16], (const float*)d_in[22]};
  const float* aw_b[2]   = {(const float*)d_in[17], (const float*)d_in[23]};
  const float* ha_b[2]   = {(const float*)d_in[18], (const float*)d_in[24]};
  const float* wa_b[2]   = {(const float*)d_in[19], (const float*)d_in[25]};

  // workspace carve-out
  char* wsb = (char*)d_ws;
  size_t off = 0;
  auto take = [&](size_t bytes) -> char* {
    off = (off + 255) & ~(size_t)255;
    char* p = wsb + off;
    off += bytes;
    return p;
  };
  bf16_t *xBF[2], *qBF[2], *kvBF[2], *qWT[2], *kvWT[2], *pWT[2];
  bf16_t *ahBF[2], *vT[2], *probsA[2], *avT[2], *p2[2], *outBF[2];
  float  *pb[2], *ab[2], *scoresA[2], *s2[2], *outPre[2];
  for (int m = 0; m < 2; ++m) {
    xBF[m]     = (bf16_t*)take(kBN * kC * 2);
    qBF[m]     = (bf16_t*)take(kBN * kC * 2);
    kvBF[m]    = (bf16_t*)take(kBN * 2 * kC * 2);
    qWT[m]     = (bf16_t*)take((long)kC * kC * 2);
    kvWT[m]    = (bf16_t*)take((long)kC * 2 * kC * 2);
    pWT[m]     = (bf16_t*)take((long)kC * kC * 2);
    ahBF[m]    = (bf16_t*)take((long)kB * kH * kAP * kD * 2);
    vT[m]      = (bf16_t*)take((long)kB * kH * kD * kNP * 2);
    pb[m]      = (float*)take((long)kH * kA * kN * 4);
    ab[m]      = (float*)take((long)kH * kN * kA * 4);
    scoresA[m] = (float*)take((long)kB * kH * kAP * kN * 4);
    probsA[m]  = (bf16_t*)take((long)kB * kH * kAP * kNP * 2);
    avT[m]     = (bf16_t*)take((long)kB * kH * kD * kAP * 2);
    s2[m]      = (float*)take((long)kB * kH * kN * kAP * 4);
    p2[m]      = (bf16_t*)take((long)kB * kH * kN * kAP * 2);
    outPre[m]  = (float*)take(kBN * kC * 4);
    outBF[m]   = (bf16_t*)take(kBN * kC * 2);
  }

  // ---------------- per-modality staging + projections ----------------
  for (int m = 0; m < 2; ++m) {
    k_cvt_bf16<<<ew(kBN * kC), 256, 0, stream>>>(fea[m], xBF[m], kBN * kC);
    k_transpose_w<<<ew((long)kC * kC), 256, 0, stream>>>(q_w[m], qWT[m], kC, kC);
    k_transpose_w<<<ew((long)kC * 2 * kC), 256, 0, stream>>>(kv_w[m], kvWT[m], kC, 2 * kC);
    k_transpose_w<<<ew((long)kC * kC), 256, 0, stream>>>(proj_w[m], pWT[m], kC, kC);

    // q = x @ q_w   (bf16 out)
    run_gemm(stream, xBF[m], kC, qWT[m], kC, qBF[m], kC, nullptr,
             (int)kBN, kC, kC, 1, 0, 0, 0, 0, 0, 0, 1.f, 1, true, false);
    // kv = x @ kv_w (bf16 out; cols [0,768)=K, [768,1536)=V)
    run_gemm(stream, xBF[m], kC, kvWT[m], kC, kvBF[m], 2 * kC, nullptr,
             (int)kBN, 2 * kC, kC, 1, 0, 0, 0, 0, 0, 0, 1.f, 1, true, false);

    k_pool_ah<<<ew((long)kB * kAP * kC), 256, 0, stream>>>(qBF[m], ahBF[m]);
    k_build_vT<<<ew((long)kB * kH * kD * kNP), 256, 0, stream>>>(kvBF[m], vT[m]);
    k_pos_bias<<<ew((long)kH * kA * kN), 256, 0, stream>>>(an_b[m], ah_b[m], aw_b[m], pb[m]);
    k_ag_bias<<<ew((long)kH * kN * kA), 256, 0, stream>>>(na_b[m], ha_b[m], wa_b[m], ab[m]);
  }

  // ---------------- cross-modal agent aggregation ----------------
  // src = modality whose K/V (and pos-bias) are attended; agents come from 1-src.
  for (int src = 0; src < 2; ++src) {
    const int ag = 1 - src;   // agent modality
    // scores[b,h,a,n] = scale * ah(ag) . K(src)
    run_gemm(stream, ahBF[ag], kD, kvBF[src], 2 * kC, scoresA[src], kN, nullptr,
             kAP, kN, kD, kH,
             (long)kH * kAP * kD, (long)kAP * kD,          // A strides
             (long)kN * 2 * kC,   kD,                      // BT: K-part col window h*64
             (long)kH * kAP * kN, (long)kAP * kN,          // C strides
             kScale, kB * kH, false, false);
    k_softmax_agent<<<dim3((unsigned)(((long)kB * kH * kAP + 3) / 4)), 128, 0, stream>>>(
        scoresA[src], pb[src], probsA[src]);
    // agent_v^T[b,h,d,a] = (probs @ V(src))^T   (bf16, transposed store)
    run_gemm(stream, probsA[src], kNP, vT[src], kNP, avT[src], kAP, nullptr,
             kAP, kD, kNP, kH,
             (long)kH * kAP * kNP, (long)kAP * kNP,
             (long)kH * kD * kNP,  (long)kD * kNP,
             (long)kH * kD * kAP,  (long)kD * kAP,
             1.f, kB * kH, true, true);
  }

  // ---------------- agent broadcast + output ----------------
  // mod=0 (rgb): q(rgb) vs agents(depth)=ahBF[1], bias ab[1], values avT[0]
  // mod=1 (depth): q(depth) vs agents(rgb)=ahBF[0], bias ab[0], values avT[1]
  for (int mod = 0; mod < 2; ++mod) {
    const int agm = 1 - mod;
    run_gemm(stream, qBF[mod], kC, ahBF[agm], kD, s2[mod], kAP, nullptr,
             kN, kAP, kD, kH,
             (long)kN * kC, kD,                             // A: head col window
             (long)kH * kAP * kD, (long)kAP * kD,
             (long)kH * kN * kAP, (long)kN * kAP,
             kScale, kB * kH, false, false);
    k_softmax_q<<<dim3((unsigned)(((long)kB * kH * kN + 3) / 4)), 128, 0, stream>>>(
        s2[mod], ab[agm], p2[mod]);
    // out[b,n,h*64+d] = p2 @ agent_v
    run_gemm(stream, p2[mod], kAP, avT[mod], kAP, outPre[mod], kC, nullptr,
             kN, kD, kAP, kH,
             (long)kH * kN * kAP, (long)kN * kAP,
             (long)kH * kD * kAP, (long)kD * kAP,
             (long)kN * kC, kD,
             1.f, kB * kH, false, false);
    // += depthwise conv of own V, convert to bf16
    k_dwc_add<<<ew(kBN * kC), 256, 0, stream>>>(kvBF[mod], dwc_w[mod], dwc_b[mod],
                                                outPre[mod], outBF[mod]);
    // final projection + bias -> d_out (f32), rgb first then depth
    float* dst = (float*)d_out + (long)mod * kBN * kC;
    run_gemm(stream, outBF[mod], kC, pWT[mod], kC, dst, kC, proj_b[mod],
             (int)mod == 0 ? (int)kBN : (int)kBN, kC, kC, 1, 0, 0, 0, 0, 0, 0,
             1.f, 1, false, false);
  }
}
